// Block_44074954391877
// MI455X (gfx1250) — matrix operand order repaired and checked by reading, not compiled, not run on hardware
//
#include <hip/hip_runtime.h>
#include <hip/hip_bf16.h>
#include <math.h>

// ---------- types ----------
typedef __bf16 bf16;
typedef __attribute__((ext_vector_type(16))) __bf16 v16bf;
typedef __attribute__((ext_vector_type(8)))  __bf16 v8bf;
typedef __attribute__((ext_vector_type(8)))  float  v8f;
typedef __attribute__((ext_vector_type(4)))  float  v4f;
typedef __attribute__((ext_vector_type(4)))  int    v4i;

// async global->LDS (CDNA5). Builtin signature (from hipcc diagnostic):
//   (int4 addrspace(1)* src, int4 addrspace(3)* dst, imm int off, imm int cpol)
#if __has_builtin(__builtin_amdgcn_global_load_async_to_lds_b128)
#define HAVE_ASYNC_LDS 1
typedef __attribute__((address_space(1))) v4i as1_v4i;
typedef __attribute__((address_space(3))) v4i as3_v4i;
__device__ __forceinline__ void async_b128(const void* g, void* l)
{
    __builtin_amdgcn_global_load_async_to_lds_b128(
        (as1_v4i*)(v4i*)(void*)g, (as3_v4i*)(v4i*)l, 0, 0);
}
#endif
#if __has_builtin(__builtin_amdgcn_s_wait_asynccnt)
#define WAIT_ASYNC(n) __builtin_amdgcn_s_wait_asynccnt(n)
#elif defined(HAVE_ASYNC_LDS)
#define WAIT_ASYNC(n) asm volatile("s_wait_asynccnt %0" ::"n"(n) : "memory")
#else
#define WAIT_ASYNC(n) ((void)0)
#endif

// ---------- problem constants ----------
#define DIMC     1024
#define HEADS    16
#define HEAD_D   64
#define HIDDEN   4096
#define BBATCH   4
#define TTIME    8
#define NPATCH   196
#define SEQS     197                 // 1 + NPATCH
#define MT       (BBATCH*NPATCH*TTIME)   // 6272 temporal tokens (b,n,t)
#define MS       (BBATCH*TTIME*SEQS)     // 6304 spatial tokens  (b,t,j)
#define MX       (BBATCH*(1+NPATCH*TTIME)) // 6276 x2 tokens
#define LNEPS    1e-5f
#define ATT_SCALE 0.125f             // 64^-0.5

// ---------- GEMM tiling ----------
#define BM 128
#define BN 128
#define BK 32
#define LDT 40   // LDS row stride in bf16 (32 + 8 pad, keeps 16B alignment)

// ---------- attention tiling ----------
#define NJT 13        // 13 key tiles of 16 -> 208
#define SEQ16 208
#define SEQ_PAD 224   // 7 chunks of 32 for the P x V contraction

// =====================================================================
// WMMA GEMM: out[M,N] = A[M,K](bf16,row) @ W[N,K](bf16,row)^T + bias
// 4 waves, block tile 128x128, wave tile 64x64 (4x4 16x16 accum frags).
// Double-buffered LDS fed by GLOBAL_LOAD_ASYNC_TO_LDS_B128 (ASYNCcnt).
// =====================================================================
__device__ __forceinline__ void
load_tile(const bf16* __restrict__ A, const bf16* __restrict__ W,
          bf16* __restrict__ lA, bf16* __restrict__ lB,
          int row0, int col0, int M, int K, int k0, int tid)
{
    // 128x32 halfs per tile = 512 16B chunks; 128 threads x 4 chunks each.
#pragma unroll
    for (int p = 0; p < 4; p++) {
        const int idx = p * 128 + tid;
        const int r   = idx >> 2;
        const int ch  = (idx & 3) * 8;          // halfs
        int gr = row0 + r; if (gr >= M) gr = M - 1;   // clamp, keep EXEC full
        const bf16* gp = A + (size_t)gr * K + k0 + ch;
        bf16*       lp = lA + r * LDT + ch;
        const bf16* wp = W + (size_t)(col0 + r) * K + k0 + ch;
        bf16*       lq = lB + r * LDT + ch;
#ifdef HAVE_ASYNC_LDS
        async_b128(gp, lp);
        async_b128(wp, lq);
#else
        *(v8bf*)lp = *(const v8bf*)gp;
        *(v8bf*)lq = *(const v8bf*)wp;
#endif
    }
}

__global__ void __launch_bounds__(128)
gemm_bf16(const bf16* __restrict__ A, const bf16* __restrict__ W,
          const float* __restrict__ bias,
          float* __restrict__ outF, bf16* __restrict__ outB,
          int M, int Nc, int K, int act)
{
    __shared__ bf16 As[2][BM * LDT];
    __shared__ bf16 Bs[2][BN * LDT];

    const int tid  = threadIdx.x;
    const int lane = tid & 31;
    const int wave = tid >> 5;     // 4 waves
    const int wm   = wave & 1;     // 0..1 : 64-row slab
    const int wn   = wave >> 1;    // 0..1 : 64-col slab
    const int row0 = blockIdx.y * BM;
    const int col0 = blockIdx.x * BN;

    const int ml   = lane & 15;    // fragment row/col within 16
    const int ksel = lane >> 4;    // K-half select per ISA 7.12.2 layout

    v8f acc[4][4];
#pragma unroll
    for (int i = 0; i < 4; i++)
#pragma unroll
        for (int j = 0; j < 4; j++)
#pragma unroll
            for (int e = 0; e < 8; e++) acc[i][j][e] = 0.f;

    // prologue: tile 0 -> buffer 0
    load_tile(A, W, As[0], Bs[0], row0, col0, M, K, 0, tid);

    int cur = 0;
    for (int k0 = 0; k0 < K; k0 += BK) {
        const bool has_next = (k0 + BK) < K;
        if (has_next)   // overlap next tile's HBM latency with this tile's WMMAs
            load_tile(A, W, As[cur ^ 1], Bs[cur ^ 1], row0, col0, M, K, k0 + BK, tid);
        if (has_next) { WAIT_ASYNC(8); } else { WAIT_ASYNC(0); }  // 8 async/wave/tile
        __syncthreads();

        // ---- fragments per ISA 7.12.2 (wave32) ----
        v16bf fa[4], fb[4];
#pragma unroll
        for (int mi = 0; mi < 4; mi++) {
            const bf16* rp = &As[cur][(wm * 64 + mi * 16 + ml) * LDT];
            v8bf lo = *(const v8bf*)(rp + ksel * 8);       // K = 8*ksel + 0..7
            v8bf hi = *(const v8bf*)(rp + 16 + ksel * 8);  // K = 16 + 8*ksel + 0..7
#pragma unroll
            for (int e = 0; e < 8; e++) { fa[mi][e] = lo[e]; fa[mi][e + 8] = hi[e]; }
        }
#pragma unroll
        for (int ni = 0; ni < 4; ni++) {
            const bf16* rp = &Bs[cur][(wn * 64 + ni * 16 + ml) * LDT];
            v8bf lo = *(const v8bf*)(rp + ksel * 8);       // K = 8*ksel + 0..7
            v8bf hi = *(const v8bf*)(rp + 16 + ksel * 8);  // K = 16 + 8*ksel + 0..7
#pragma unroll
            for (int e = 0; e < 8; e++) { fb[ni][e] = lo[e]; fb[ni][e + 8] = hi[e]; }
        }
#pragma unroll
        for (int mi = 0; mi < 4; mi++)
#pragma unroll
            for (int ni = 0; ni < 4; ni++)
                acc[mi][ni] = __builtin_amdgcn_wmma_f32_16x16x32_bf16(
                    false, fa[mi], false, fb[ni], (short)0, acc[mi][ni],
                    false, false);
        __syncthreads();   // protect buf[cur] before it is refilled
        cur ^= 1;
    }

    // ---- epilogue: C layout lane(l): N = l&15, M = vgpr + 8*(l>>4) ----
#pragma unroll
    for (int mi = 0; mi < 4; mi++) {
        const int rbase = row0 + wm * 64 + mi * 16 + ksel * 8;
#pragma unroll
        for (int ni = 0; ni < 4; ni++) {
            const int col = col0 + wn * 64 + ni * 16 + ml;
            const float bv = bias ? bias[col] : 0.f;
#pragma unroll
            for (int v = 0; v < 8; v++) {
                const int r = rbase + v;
                if (r < M) {
                    float val = acc[mi][ni][v] + bv;
                    if (act == 1)
                        val = 0.5f * val * (1.f + erff(val * 0.70710678118654752f));
                    const size_t o = (size_t)r * Nc + col;
                    if (outF) outF[o] = val;
                    if (outB) outB[o] = (bf16)val;
                }
            }
        }
    }
}

// =====================================================================
// LayerNorm over 1024 (one block per token, 256 threads, 4 elems each)
// =====================================================================
__device__ __forceinline__ void ln_row(const float* __restrict__ src,
                                       const float* __restrict__ g,
                                       const float* __restrict__ b,
                                       bf16* __restrict__ dst)
{
    __shared__ float red[16];
    const int tid = threadIdx.x;
    const v4f xv = *(const v4f*)(src + tid * 4);
    float s  = xv[0] + xv[1] + xv[2] + xv[3];
    float s2 = xv[0]*xv[0] + xv[1]*xv[1] + xv[2]*xv[2] + xv[3]*xv[3];
#pragma unroll
    for (int off = 16; off; off >>= 1) {
        s  += __shfl_xor(s,  off, 32);
        s2 += __shfl_xor(s2, off, 32);
    }
    if ((tid & 31) == 0) { red[tid >> 5] = s; red[8 + (tid >> 5)] = s2; }
    __syncthreads();
    if (tid == 0) {
        float ts = 0.f, ts2 = 0.f;
#pragma unroll
        for (int i = 0; i < 8; i++) { ts += red[i]; ts2 += red[8 + i]; }
        const float mean = ts * (1.f / DIMC);
        const float var  = ts2 * (1.f / DIMC) - mean * mean;
        red[0] = mean; red[1] = rsqrtf(var + LNEPS);
    }
    __syncthreads();
    const float mean = red[0], inv = red[1];
#pragma unroll
    for (int e = 0; e < 4; e++) {
        const int c = tid * 4 + e;
        dst[c] = (bf16)((xv[e] - mean) * inv * g[c] + b[c]);
    }
}

// temporal: token m=(b*N+n)*T+t reads x[b][t][n][:]
__global__ void __launch_bounds__(256)
ln_temporal_k(const float* __restrict__ x, const float* __restrict__ g,
              const float* __restrict__ b, bf16* __restrict__ out)
{
    const int m = blockIdx.x;
    const int bi  = m / (NPATCH * TTIME);
    const int rem = m % (NPATCH * TTIME);
    const int n = rem / TTIME, t = rem % TTIME;
    const float* src = x + (((size_t)bi * TTIME + t) * NPATCH + n) * DIMC;
    ln_row(src, g, b, out + (size_t)m * DIMC);
}

// spatial: token s=(b*T+t)*197+j; j==0 -> cls_token[b], else xt[(b*N+j-1)*T+t]
__global__ void __launch_bounds__(256)
ln_spatial_k(const float* __restrict__ xt, const float* __restrict__ cls,
             const float* __restrict__ g, const float* __restrict__ b,
             bf16* __restrict__ out)
{
    const int s  = blockIdx.x;
    const int bt = s / SEQS, j = s % SEQS;
    const int bi = bt / TTIME, t = bt % TTIME;
    const float* src = (j == 0)
        ? cls + (size_t)bi * DIMC
        : xt + (((size_t)bi * NPATCH + (j - 1)) * TTIME + t) * DIMC;
    ln_row(src, g, b, out + (size_t)s * DIMC);
}

__global__ void __launch_bounds__(256)
ln_plain_k(const float* __restrict__ in, const float* __restrict__ g,
           const float* __restrict__ b, bf16* __restrict__ out)
{
    ln_row(in + (size_t)blockIdx.x * DIMC, g, b,
           out + (size_t)blockIdx.x * DIMC);
}

// =====================================================================
// Temporal attention: one block per (b,n), 16 heads, seq=8, all fp32.
// (negligible FLOPs; VALU is fine here)
// =====================================================================
__global__ void __launch_bounds__(256)
attn_temporal_k(const float* __restrict__ QKV, bf16* __restrict__ O)
{
    __shared__ float qs[TTIME][HEAD_D], ks[TTIME][HEAD_D], vs[TTIME][HEAD_D];
    __shared__ float ps[TTIME][TTIME];
    const int bn  = blockIdx.x;
    const int tid = threadIdx.x;
    for (int h = 0; h < HEADS; h++) {
        for (int idx = tid; idx < TTIME * HEAD_D; idx += 256) {
            const int t = idx >> 6, d = idx & 63;
            const size_t base = ((size_t)(bn * TTIME + t)) * (3 * DIMC) + h * HEAD_D + d;
            qs[t][d] = QKV[base];
            ks[t][d] = QKV[base + DIMC];
            vs[t][d] = QKV[base + 2 * DIMC];
        }
        __syncthreads();
        if (tid < TTIME * TTIME) {
            const int i = tid >> 3, j = tid & 7;
            float dot = 0.f;
#pragma unroll
            for (int d = 0; d < HEAD_D; d++) dot += qs[i][d] * ks[j][d];
            ps[i][j] = dot * ATT_SCALE;
        }
        __syncthreads();
        if (tid < TTIME) {
            float mx = ps[tid][0];
#pragma unroll
            for (int j = 1; j < TTIME; j++) mx = fmaxf(mx, ps[tid][j]);
            float sum = 0.f;
#pragma unroll
            for (int j = 0; j < TTIME; j++) { float e = expf(ps[tid][j] - mx); ps[tid][j] = e; sum += e; }
            const float inv = 1.f / sum;
#pragma unroll
            for (int j = 0; j < TTIME; j++) ps[tid][j] *= inv;
        }
        __syncthreads();
        for (int idx = tid; idx < TTIME * HEAD_D; idx += 256) {
            const int t = idx >> 6, d = idx & 63;
            float o = 0.f;
#pragma unroll
            for (int j = 0; j < TTIME; j++) o += ps[t][j] * vs[j][d];
            O[((size_t)(bn * TTIME + t)) * DIMC + h * HEAD_D + d] = (bf16)o;
        }
        __syncthreads();
    }
}

// =====================================================================
// Spatial WMMA attention: one block (4 waves) per (bt, head).
// K (bf16 [j][64]) and V^T (bf16 [d][224]) staged in LDS; Q converted
// fp32->bf16 on the fly; softmax in fp32 on C-fragment layout with
// shfl_xor butterflies + tiny LDS cross-wave reductions.  62.9KB LDS.
// =====================================================================
__global__ void __launch_bounds__(128)
attn_spatial_wmma_k(const float* __restrict__ QKV, bf16* __restrict__ O)
{
    __shared__ bf16 Kl[SEQ16 * HEAD_D];      // 26.0 KB  [j][d], j>=197 zero
    __shared__ bf16 Vt[HEAD_D * SEQ_PAD];    // 28.0 KB  [d][j]
    __shared__ bf16 Pl[16 * SEQ_PAD];        //  7.0 KB  [q_local][j]
    __shared__ float redmx[4][16];
    __shared__ float redsm[4][16];

    const int bt   = blockIdx.x >> 4;
    const int h    = blockIdx.x & 15;
    const int tid  = threadIdx.x;
    const int lane = tid & 31;
    const int w    = tid >> 5;     // wave id: QK key-tile slice + AV d-slab
    const int ml   = lane & 15;
    const int ksel = lane >> 4;

    const float* Qg = QKV + ((size_t)bt * SEQS) * (3 * DIMC) + h * HEAD_D;
    const float* Kg = Qg + DIMC;
    const float* Vg = Qg + 2 * DIMC;

    // ---- stage K (rows >= 197 zeroed) ----
    for (int ch = tid; ch < SEQ16 * 16; ch += 128) {
        const int j = ch >> 4, c4 = (ch & 15) * 4;
        bf16* dst = &Kl[j * HEAD_D + c4];
        if (j < SEQS) {
            const v4f kv = *(const v4f*)(Kg + (size_t)j * (3 * DIMC) + c4);
            dst[0] = (bf16)kv[0]; dst[1] = (bf16)kv[1];
            dst[2] = (bf16)kv[2]; dst[3] = (bf16)kv[3];
        } else {
            dst[0] = (bf16)0.f; dst[1] = (bf16)0.f;
            dst[2] = (bf16)0.f; dst[3] = (bf16)0.f;
        }
    }
    // ---- stage V^T (row-clamped; P=0 masks j>=197) ----
    for (int ch = tid; ch < SEQ_PAD * 16; ch += 128) {
        const int j = ch >> 4, c4 = (ch & 15) * 4;
        const int jj = j < SEQS ? j : (SEQS - 1);
        const v4f vv = *(const v4f*)(Vg + (size_t)jj * (3 * DIMC) + c4);
        Vt[(c4 + 0) * SEQ_PAD + j] = (bf16)vv[0];
        Vt[(c4 + 1) * SEQ_PAD + j] = (bf16)vv[1];
        Vt[(c4 + 2) * SEQ_PAD + j] = (bf16)vv[2];
        Vt[(c4 + 3) * SEQ_PAD + j] = (bf16)vv[3];
    }
    // ---- zero P pad columns 208..223 once ----
    {
        const int r = tid >> 3, c = SEQ16 + (tid & 7) * 2;
        Pl[r * SEQ_PAD + c]     = (bf16)0.f;
        Pl[r * SEQ_PAD + c + 1] = (bf16)0.f;
    }
    __syncthreads();

    for (int qt = 0; qt < NJT; qt++) {
        // ---- Q A-fragments (fp32 -> bf16, row-clamped) ----
        const int mq0 = qt * 16 + ml;
        const int mq  = mq0 < SEQS ? mq0 : (SEQS - 1);
        const float* Qrow = Qg + (size_t)mq * (3 * DIMC);
        v16bf qa[2];
#pragma unroll
        for (int c = 0; c < 2; c++) {
            const float* p0 = Qrow + c * 32 + ksel * 8;
            const float* p1 = p0 + 16;
            const v4f a0 = *(const v4f*)p0, a1 = *(const v4f*)(p0 + 4);
            const v4f b0 = *(const v4f*)p1, b1 = *(const v4f*)(p1 + 4);
#pragma unroll
            for (int e = 0; e < 4; e++) {
                qa[c][e]      = (bf16)a0[e];
                qa[c][e + 4]  = (bf16)a1[e];
                qa[c][e + 8]  = (bf16)b0[e];
                qa[c][e + 12] = (bf16)b1[e];
            }
        }
        // ---- S = (Q K^T) * scale for this wave's key tiles ----
        v8f S[4];
#pragma unroll
        for (int i = 0; i < 4; i++) {
            const int jt = w + i * 4;
            if (jt < NJT) {
                v8f s;
#pragma unroll
                for (int e = 0; e < 8; e++) s[e] = 0.f;
#pragma unroll
                for (int c = 0; c < 2; c++) {
                    const bf16* kr = &Kl[(jt * 16 + ml) * HEAD_D + c * 32 + ksel * 8];
                    const v8bf lo = *(const v8bf*)kr, hi = *(const v8bf*)(kr + 16);
                    v16bf kb;
#pragma unroll
                    for (int e = 0; e < 8; e++) { kb[e] = lo[e]; kb[e + 8] = hi[e]; }
                    s = __builtin_amdgcn_wmma_f32_16x16x32_bf16(
                        false, qa[c], false, kb, (short)0, s, false, false);
                }
                const int jcol = jt * 16 + ml;
#pragma unroll
                for (int e = 0; e < 8; e++)
                    s[e] = (jcol < SEQS) ? s[e] * ATT_SCALE : -1e30f;
                S[i] = s;
            }
        }
        // ---- row max: local -> 16-lane butterfly -> cross-wave via LDS ----
        float mloc[8];
#pragma unroll
        for (int v = 0; v < 8; v++) {
            float mv = -1e30f;
#pragma unroll
            for (int i = 0; i < 4; i++)
                if (w + i * 4 < NJT) mv = fmaxf(mv, S[i][v]);
            mloc[v] = mv;
        }
#pragma unroll
        for (int off = 1; off < 16; off <<= 1)
#pragma unroll
            for (int v = 0; v < 8; v++)
                mloc[v] = fmaxf(mloc[v], __shfl_xor(mloc[v], off, 32));
        if (ml == 0)
#pragma unroll
            for (int v = 0; v < 8; v++) redmx[w][ksel * 8 + v] = mloc[v];
        __syncthreads();
        float gmax[8];
#pragma unroll
        for (int v = 0; v < 8; v++) {
            float mv = fmaxf(redmx[0][ksel * 8 + v], redmx[1][ksel * 8 + v]);
            mv = fmaxf(mv, redmx[2][ksel * 8 + v]);
            gmax[v] = fmaxf(mv, redmx[3][ksel * 8 + v]);
        }
        // ---- exp, row sum, write P (bf16) ----
        float sloc[8];
#pragma unroll
        for (int v = 0; v < 8; v++) sloc[v] = 0.f;
#pragma unroll
        for (int i = 0; i < 4; i++) {
            const int jt = w + i * 4;
            if (jt < NJT) {
#pragma unroll
                for (int v = 0; v < 8; v++) {
                    const float p = expf(S[i][v] - gmax[v]);  // masked cols -> 0
                    sloc[v] += p;
                    Pl[(ksel * 8 + v) * SEQ_PAD + jt * 16 + ml] = (bf16)p;
                }
            }
        }
#pragma unroll
        for (int off = 1; off < 16; off <<= 1)
#pragma unroll
            for (int v = 0; v < 8; v++)
                sloc[v] += __shfl_xor(sloc[v], off, 32);
        if (ml == 0)
#pragma unroll
            for (int v = 0; v < 8; v++) redsm[w][ksel * 8 + v] = sloc[v];
        __syncthreads();   // P + partial sums visible to all waves
        float rinv[8];
#pragma unroll
        for (int v = 0; v < 8; v++)
            rinv[v] = 1.f / (redsm[0][ksel * 8 + v] + redsm[1][ksel * 8 + v]
                           + redsm[2][ksel * 8 + v] + redsm[3][ksel * 8 + v]);
        // ---- O(d-slab = wave) = P(16x224) x V(224x16) ----
        v8f o;
#pragma unroll
        for (int e = 0; e < 8; e++) o[e] = 0.f;
#pragma unroll
        for (int c7 = 0; c7 < 7; c7++) {
            const bf16* pr = &Pl[ml * SEQ_PAD + c7 * 32];
            const v8bf plo = *(const v8bf*)(pr + ksel * 8);
            const v8bf phi = *(const v8bf*)(pr + 16 + ksel * 8);
            v16bf pa;
#pragma unroll
            for (int e = 0; e < 8; e++) { pa[e] = plo[e]; pa[e + 8] = phi[e]; }
            const bf16* vr = &Vt[(w * 16 + ml) * SEQ_PAD + c7 * 32 + ksel * 8];
            const v8bf vlo = *(const v8bf*)vr, vhi = *(const v8bf*)(vr + 16);
            v16bf vb;
#pragma unroll
            for (int e = 0; e < 8; e++) { vb[e] = vlo[e]; vb[e + 8] = vhi[e]; }
            o = __builtin_amdgcn_wmma_f32_16x16x32_bf16(
                false, pa, false, vb, (short)0, o, false, false);
        }
#pragma unroll
        for (int v = 0; v < 8; v++) {
            const int q = qt * 16 + ksel * 8 + v;
            if (q < SEQS)
                O[((size_t)(bt * SEQS + q)) * DIMC + h * HEAD_D + w * 16 + ml]
                    = (bf16)(o[v] * rinv[v]);
        }
        __syncthreads();   // Pl reused next q-tile
    }
}

// =====================================================================
// glue kernels
// =====================================================================
__global__ void __launch_bounds__(256)
cvt_f32_bf16_k(const float* __restrict__ in, bf16* __restrict__ out, int n)
{
    const int i = blockIdx.x * 256 + threadIdx.x;
    if (i < n) out[i] = (bf16)in[i];
}

__global__ void __launch_bounds__(256)
qkv_bias_k(const float* __restrict__ qb, const float* __restrict__ vb,
           float* __restrict__ out)
{
    const int i = blockIdx.x * 256 + threadIdx.x;   // 0..3071
    out[i] = (i < DIMC) ? qb[i] : (i < 2 * DIMC ? 0.f : vb[i - 2 * DIMC]);
}

// xt[m] = F[m] + x[b][t][n]   (m = (b*N+n)*T+t)
__global__ void __launch_bounds__(256)
resid_xt_k(const float* __restrict__ F, const float* __restrict__ x,
           float* __restrict__ xt)
{
    const size_t i = (size_t)blockIdx.x * 256 + threadIdx.x;
    const int m = (int)(i / DIMC), c = (int)(i % DIMC);
    const int bi  = m / (NPATCH * TTIME);
    const int rem = m % (NPATCH * TTIME);
    const int n = rem / TTIME, t = rem % TTIME;
    xt[i] = F[i] + x[(((size_t)bi * TTIME + t) * NPATCH + n) * DIMC + c];
}

// x2[b][0] = cls[b] + mean_t Ps[(b*T+t)*197+0]
// x2[b][1+n*T+t] = xt[(b*N+n)*T+t] + Ps[(b*T+t)*197+1+n]
__global__ void __launch_bounds__(256)
assemble_x2_k(const float* __restrict__ xt, const float* __restrict__ cls,
              const float* __restrict__ Ps, float* __restrict__ x2)
{
    const size_t i = (size_t)blockIdx.x * 256 + threadIdx.x;
    const int m = (int)(i / DIMC), c = (int)(i % DIMC);
    const int bi = m / (1 + NPATCH * TTIME);
    const int r  = m % (1 + NPATCH * TTIME);
    float val;
    if (r == 0) {
        float s = 0.f;
#pragma unroll
        for (int t = 0; t < TTIME; t++)
            s += Ps[((size_t)(bi * TTIME + t) * SEQS) * DIMC + c];
        val = cls[(size_t)bi * DIMC + c] + s * (1.f / TTIME);
    } else {
        const int nt = r - 1, n = nt / TTIME, t = nt % TTIME;
        val = xt[(((size_t)bi * NPATCH + n) * TTIME + t) * DIMC + c]
            + Ps[((size_t)(bi * TTIME + t) * SEQS + 1 + n) * DIMC + c];
    }
    x2[i] = val;
}

// out = [xo (B,T,N,C) flat][cls_out (B,C)]
__global__ void __launch_bounds__(256)
final_out_k(const float* __restrict__ x2, const float* __restrict__ H2,
            float* __restrict__ out)
{
    const size_t i = (size_t)blockIdx.x * 256 + threadIdx.x;
    const int m = (int)(i / DIMC), c = (int)(i % DIMC);
    const float val = x2[i] + H2[i];
    const int bi = m / (1 + NPATCH * TTIME);
    const int r  = m % (1 + NPATCH * TTIME);
    if (r == 0) {
        out[(size_t)BBATCH * TTIME * NPATCH * DIMC + (size_t)bi * DIMC + c] = val;
    } else {
        const int nt = r - 1, n = nt / TTIME, t = nt % TTIME;
        out[(((size_t)(bi * TTIME + t)) * NPATCH + n) * DIMC + c] = val;
    }
}

// =====================================================================
// host launcher
// =====================================================================
extern "C" void kernel_launch(void* const* d_in, const int* in_sizes, int n_in,
                              void* d_out, int out_size, void* d_ws, size_t ws_size,
                              hipStream_t stream)
{
    (void)in_sizes; (void)n_in; (void)out_size; (void)ws_size;

    const float* x      = (const float*)d_in[0];
    const float* cls    = (const float*)d_in[1];
    const float* tln_g  = (const float*)d_in[2];
    const float* tln_b  = (const float*)d_in[3];
    const float* t_wqkv = (const float*)d_in[4];
    const float* t_qb   = (const float*)d_in[5];
    const float* t_vb   = (const float*)d_in[6];
    const float* t_wproj= (const float*)d_in[7];
    const float* t_bproj= (const float*)d_in[8];
    const float* t_fc_w = (const float*)d_in[9];
    const float* t_fc_b = (const float*)d_in[10];
    const float* n1_g   = (const float*)d_in[11];
    const float* n1_b   = (const float*)d_in[12];
    const float* s_wqkv = (const float*)d_in[13];
    const float* s_qb   = (const float*)d_in[14];
    const float* s_vb   = (const float*)d_in[15];
    const float* s_wproj= (const float*)d_in[16];
    const float* s_bproj= (const float*)d_in[17];
    const float* n2_g   = (const float*)d_in[18];
    const float* n2_b   = (const float*)d_in[19];
    const float* fc1_w  = (const float*)d_in[20];
    const float* fc1_b  = (const float*)d_in[21];
    const float* fc2_w  = (const float*)d_in[22];
    const float* fc2_b  = (const float*)d_in[23];
    float* out = (float*)d_out;

    // ---- workspace layout ----
    char* ws = (char*)d_ws;
    size_t off = 0;
    auto alloc = [&](size_t bytes) -> char* {
        char* p = ws + off;
        off = (off + bytes + 255) & ~(size_t)255;
        return p;
    };
    bf16* wq_t  = (bf16*)alloc((size_t)3 * DIMC * DIMC * 2);
    bf16* wp_t  = (bf16*)alloc((size_t)DIMC * DIMC * 2);
    bf16* wfc_t = (bf16*)alloc((size_t)DIMC * DIMC * 2);
    bf16* wq_s  = (bf16*)alloc((size_t)3 * DIMC * DIMC * 2);
    bf16* wp_s  = (bf16*)alloc((size_t)DIMC * DIMC * 2);
    bf16* w1    = (bf16*)alloc((size_t)HIDDEN * DIMC * 2);
    bf16* w2    = (bf16*)alloc((size_t)DIMC * HIDDEN * 2);
    float* biasT = (float*)alloc((size_t)3 * DIMC * 4);
    float* biasS = (float*)alloc((size_t)3 * DIMC * 4);
    bf16*  Aln  = (bf16*) alloc((size_t)MS * DIMC * 2);     // LN out (reused)
    float* QKV  = (float*)alloc((size_t)MS * 3 * DIMC * 4); // qkv (reused)
    bf16*  Obf  = (bf16*) alloc((size_t)MS * DIMC * 2);     // attn out (reused)
    float* Pf   = (float*)alloc((size_t)MS * DIMC * 4);     // proj out (reused)
    bf16*  Pbf  = (bf16*) alloc((size_t)MT * DIMC * 2);     // proj out bf16 (temporal)
    float* Ff   = (float*)alloc((size_t)MT * DIMC * 4);     // t_fc out
    float* xt   = (float*)alloc((size_t)MT * DIMC * 4);
    float* x2   = (float*)alloc((size_t)MX * DIMC * 4);
    bf16*  Hbf  = (bf16*) alloc((size_t)MX * HIDDEN * 2);   // gelu(fc1) bf16
    float* H2   = (float*)alloc((size_t)MX * DIMC * 4);

    // ---- weight conversion ----
    auto cvt = [&](const float* src, bf16* dst, int n) {
        cvt_f32_bf16_k<<<(n + 255) / 256, 256, 0, stream>>>(src, dst, n);
    };
    cvt(t_wqkv, wq_t, 3 * DIMC * DIMC);
    cvt(t_wproj, wp_t, DIMC * DIMC);
    cvt(t_fc_w, wfc_t, DIMC * DIMC);
    cvt(s_wqkv, wq_s, 3 * DIMC * DIMC);
    cvt(s_wproj, wp_s, DIMC * DIMC);
    cvt(fc1_w, w1, HIDDEN * DIMC);
    cvt(fc2_w, w2, DIMC * HIDDEN);
    qkv_bias_k<<<(3 * DIMC) / 256, 256, 0, stream>>>(t_qb, t_vb, biasT);
    qkv_bias_k<<<(3 * DIMC) / 256, 256, 0, stream>>>(s_qb, s_vb, biasS);

    auto gemm = [&](const bf16* A, const bf16* W, const float* bias,
                    float* oF, bf16* oB, int M, int N, int K, int act) {
        dim3 grid(N / BN, (M + BM - 1) / BM);
        gemm_bf16<<<grid, 128, 0, stream>>>(A, W, bias, oF, oB, M, N, K, act);
    };

    // ---- temporal branch ----
    ln_temporal_k<<<MT, 256, 0, stream>>>(x, tln_g, tln_b, Aln);
    gemm(Aln, wq_t, biasT, QKV, nullptr, MT, 3 * DIMC, DIMC, 0);
    attn_temporal_k<<<BBATCH * NPATCH, 256, 0, stream>>>(QKV, Obf);
    gemm(Obf, wp_t, t_bproj, Pf, Pbf, MT, DIMC, DIMC, 0);
    gemm(Pbf, wfc_t, t_fc_b, Ff, nullptr, MT, DIMC, DIMC, 0);
    resid_xt_k<<<(MT * DIMC) / 256, 256, 0, stream>>>(Ff, x, xt);

    // ---- spatial branch ----
    ln_spatial_k<<<MS, 256, 0, stream>>>(xt, cls, n1_g, n1_b, Aln);
    gemm(Aln, wq_s, biasS, QKV, nullptr, MS, 3 * DIMC, DIMC, 0);
    attn_spatial_wmma_k<<<BBATCH * TTIME * HEADS, 128, 0, stream>>>(QKV, Obf);
    gemm(Obf, wp_s, s_bproj, Pf, nullptr, MS, DIMC, DIMC, 0);
    assemble_x2_k<<<(MX * DIMC) / 256, 256, 0, stream>>>(xt, cls, Pf, x2);

    // ---- MLP ----
    ln_plain_k<<<MX, 256, 0, stream>>>(x2, n2_g, n2_b, Aln);
    gemm(Aln, w1, fc1_b, nullptr, Hbf, MX, HIDDEN, DIMC, 1);  // exact GELU
    gemm(Hbf, w2, fc2_b, H2, nullptr, MX, DIMC, HIDDEN, 0);
    final_out_k<<<(MX * DIMC) / 256, 256, 0, stream>>>(x2, H2, out);
}
